// Attribute_78185584656630
// MI455X (gfx1250) — matrix operand order repaired and checked by reading, not compiled, not run on hardware
//
#include <hip/hip_runtime.h>
#include <hip/hip_bf16.h>

// ---------------------------------------------------------------------------
// MI455X (gfx1250) pipeline for the heterogeneous graph-attention reference.
// Dominant cost: two N=8192, D=128 single-head attentions (~69 GFLOP),
// L2-resident -> matrix-unit bound -> flash attention on v_wmma_f32_16x16x32_f16.
// All WMMA fragments are loaded as contiguous b128 LDS vectors (A and B^T
// layouts), weights staged transposed so no strided 16-bit LDS loads remain.
// ---------------------------------------------------------------------------

#define NN 8192
#define MM 16384
#define DD 128
#define SS 32
#define EE 131072
#define TT 2

typedef _Float16 h16 __attribute__((ext_vector_type(16)));
typedef _Float16 h8v __attribute__((ext_vector_type(8)));
typedef float    f8  __attribute__((ext_vector_type(8)));

// ---- WMMA fragment helpers (ISA 7.12.2 layouts, wave32) -------------------
// A 16x32 f16 tile, row-major in LDS, stride lda (halfs). Per lane the A
// layout is two contiguous 8-half runs: K in [8h,8h+8) and [16+8h,16+8h+8).
__device__ __forceinline__ h16 load_a(const _Float16* A, int lda, int lane) {
  int m = lane & 15, hh = lane >> 4;
  const _Float16* p = A + m * lda + 8 * hh;
  h8v lo = *(const h8v*)p;
  h8v hi = *(const h8v*)(p + 16);
  return __builtin_shufflevector(lo, hi, 0, 1, 2, 3, 4, 5, 6, 7,
                                 8, 9, 10, 11, 12, 13, 14, 15);
}

// B 32x16 tile given B^T stored row-major (Bt[n][k], stride ldb): per lane
// two contiguous 8-half runs at row n: K in [8h,8h+8) and [16+8h,16+8h+8).
__device__ __forceinline__ h16 load_bt(const _Float16* Bt, int ldb, int lane) {
  int n = lane & 15, hh = lane >> 4;
  const _Float16* p = Bt + n * ldb + 8 * hh;
  h8v lo = *(const h8v*)p;
  h8v hi = *(const h8v*)(p + 16);
  return __builtin_shufflevector(lo, hi, 0, 1, 2, 3, 4, 5, 6, 7,
                                 8, 9, 10, 11, 12, 13, 14, 15);
}

__device__ __forceinline__ f8 wmma(h16 a, h16 b, f8 c) {
  return __builtin_amdgcn_wmma_f32_16x16x32_f16(false, a, false, b,
                                                (short)0, c, false, false);
}

// ---- staging helpers (256 threads, 128x128 tiles) -------------------------
__device__ __forceinline__ void copy_tile_h16(_Float16* __restrict__ dst,
                                              const _Float16* __restrict__ src,
                                              int tid) {
  const uint4* s = (const uint4*)src;
  uint4* d = (uint4*)dst;
#pragma unroll
  for (int i = tid; i < 2048; i += 256) d[i] = s[i];  // 8 halfs per uint4
}

__device__ __forceinline__ void convert_tile_f32_h16(_Float16* __restrict__ dst,
                                                     const float* __restrict__ src,
                                                     int tid) {
#pragma unroll
  for (int i = tid; i < 2048; i += 256) {
    float4 a = ((const float4*)src)[2 * i];
    float4 b = ((const float4*)src)[2 * i + 1];
    h8v h;
    h[0] = (_Float16)a.x; h[1] = (_Float16)a.y;
    h[2] = (_Float16)a.z; h[3] = (_Float16)a.w;
    h[4] = (_Float16)b.x; h[5] = (_Float16)b.y;
    h[6] = (_Float16)b.z; h[7] = (_Float16)b.w;
    ((h8v*)dst)[i] = h;
  }
}

// W row-major [k][n] f32 -> LDS B^T [n][k] f16 (contiguous LDS writes)
__device__ __forceinline__ void stage_wT(_Float16* __restrict__ dst,
                                         const float* __restrict__ W, int tid) {
  for (int idx = tid; idx < 128 * 128; idx += 256)
    dst[idx] = (_Float16)W[((idx & 127) << 7) + (idx >> 7)];
}

// ---- Kernel 1: intra-type neighbor attention + residual -------------------
__global__ void intra_kernel(const float* __restrict__ x0,
                             const float* __restrict__ xh,
                             const float* __restrict__ attv,
                             const int* __restrict__ nei,
                             float* __restrict__ one,
                             _Float16* __restrict__ one_h) {
  __shared__ __align__(16) float ne[SS * DD];
  __shared__ float sc[SS];
  __shared__ float xr[DD];
  __shared__ int   ni[SS];
  __shared__ float sref_s;
  int n = blockIdx.x, tid = threadIdx.x;
  if (tid < SS) ni[tid] = nei[n * SS + tid];
  xr[tid] = x0[(long)n * DD + tid];
  __syncthreads();
  // stage 32 gathered rows, float4 granularity (32 chunks per row)
  for (int idx = tid; idx < SS * DD / 4; idx += 128) {
    int r = idx >> 5, c = idx & 31;
    ((float4*)ne)[idx] = ((const float4*)&xh[(long)ni[r] * DD])[c];
  }
  __syncthreads();
  int wave = tid >> 5, lane = tid & 31;
  if (wave == 0) {
    float p = 0.f;
#pragma unroll
    for (int j = 0; j < 4; ++j) p += xr[lane * 4 + j] * attv[lane * 4 + j];
    for (int m = 16; m >= 1; m >>= 1) p += __shfl_xor(p, m, 32);
    if (lane == 0) sref_s = p;
  }
  for (int t = 0; t < 8; ++t) {
    int s = wave + 4 * t;
    float4 nv = ((const float4*)&ne[s * DD])[lane];
    float p = nv.x * attv[DD + lane * 4] + nv.y * attv[DD + lane * 4 + 1] +
              nv.z * attv[DD + lane * 4 + 2] + nv.w * attv[DD + lane * 4 + 3];
    for (int m = 16; m >= 1; m >>= 1) p += __shfl_xor(p, m, 32);
    if (lane == 0) sc[s] = p;
  }
  __syncthreads();
  float sref = sref_s, mx = -1e30f;
  float w[SS];
#pragma unroll
  for (int s = 0; s < SS; ++s) {
    float a = sref + sc[s];
    a = (a > 0.f) ? a : 0.01f * a;  // leaky_relu(0.01)
    w[s] = a;
    mx = fmaxf(mx, a);
  }
  float sum = 0.f;
#pragma unroll
  for (int s = 0; s < SS; ++s) { w[s] = __expf(w[s] - mx); sum += w[s]; }
  float inv = 1.f / sum;
  float acc = 0.f;
#pragma unroll
  for (int s = 0; s < SS; ++s) acc += w[s] * ne[s * DD + tid];
  float v = acc * inv + xr[tid];
  one[(long)n * DD + tid]   = v;
  one_h[(long)n * DD + tid] = (_Float16)v;
}

// ---- Kernel 2: dual GEMM  O1 = A@W1, O2 = A@W2  (B^T staged) --------------
__global__ void gemm_nn_dual(const _Float16* __restrict__ Ah,
                             const float* __restrict__ W1,
                             const float* __restrict__ W2,
                             float* __restrict__ O1, float* __restrict__ O2) {
  __shared__ __align__(16) _Float16 As[128 * 128];
  __shared__ __align__(16) _Float16 Bs[128 * 128];
  int tid = threadIdx.x, wave = tid >> 5, lane = tid & 31;
  int m0 = blockIdx.x * 128;
  copy_tile_h16(As, Ah + (size_t)m0 * 128, tid);
  const float* Ws[2] = {W1, W2};
  float* Os[2] = {O1, O2};
  for (int p = 0; p < 2; ++p) {
    __syncthreads();
    stage_wT(Bs, Ws[p], tid);
    __syncthreads();
    h16 afr[4];
#pragma unroll
    for (int k = 0; k < 4; ++k)
      afr[k] = load_a(&As[(wave * 16) * 128 + k * 32], 128, lane);
    for (int nt = 0; nt < 8; ++nt) {
      f8 acc = {};
#pragma unroll
      for (int k = 0; k < 4; ++k)
        acc = wmma(afr[k], load_bt(&Bs[(nt * 16) * 128 + k * 32], 128, lane), acc);
      int col = nt * 16 + (lane & 15);
#pragma unroll
      for (int v = 0; v < 8; ++v) {
        int row = m0 + wave * 16 + v + 8 * (lane >> 4);
        Os[p][(long)row * 128 + col] = acc[v];
      }
    }
  }
}

// ---- Kernel 3: broadcast bias init ----------------------------------------
__global__ void init_bias(float* __restrict__ out, const float* __restrict__ b) {
  long i = (long)blockIdx.x * blockDim.x + threadIdx.x;  // float4 index
  if (i < (long)NN * DD / 4)
    ((float4*)out)[i] = ((const float4*)b)[i & 31];
}

// ---- Kernel 4: COO spmm scatter: out[rows[e]] += vals[e]*X[cols[e]] -------
__global__ void spmm_scatter(const int* __restrict__ rows,
                             const int* __restrict__ cols,
                             const float* __restrict__ vals,
                             const float* __restrict__ X,
                             float* __restrict__ out) {
  long gid = (long)blockIdx.x * blockDim.x + threadIdx.x;
  int e = (int)(gid >> 5);
  if (e >= EE) return;
  int lane = (int)(gid & 31);
  int r = rows[e], c = cols[e];
  float v = vals[e];
  const float4 xv = *(const float4*)&X[(long)c * DD + lane * 4];
  float* o = &out[(long)r * DD + lane * 4];
  atomicAdd(o + 0, v * xv.x);
  atomicAdd(o + 1, v * xv.y);
  atomicAdd(o + 2, v * xv.z);
  atomicAdd(o + 3, v * xv.w);
}

// ---- Kernel 5: elu + f16 convert (8 elems/thread) -------------------------
__global__ void elu_to_h16(const float* __restrict__ in, _Float16* __restrict__ out) {
  long i = (long)blockIdx.x * blockDim.x + threadIdx.x;
  if (i >= (long)NN * DD / 8) return;
  float4 a = ((const float4*)in)[2 * i];
  float4 b = ((const float4*)in)[2 * i + 1];
  float e[8] = {a.x, a.y, a.z, a.w, b.x, b.y, b.z, b.w};
  h8v h;
#pragma unroll
  for (int j = 0; j < 8; ++j)
    h[j] = (_Float16)((e[j] > 0.f) ? e[j] : (__expf(e[j]) - 1.f));
  ((h8v*)out)[i] = h;
}

// ---- Kernel 6: in_proj -> q (pre-scaled), k row-major; v transposed -------
__global__ void inproj_kernel(const _Float16* __restrict__ Hh,
                              const float* __restrict__ wi,
                              const float* __restrict__ bi,
                              _Float16* __restrict__ qh,
                              _Float16* __restrict__ kh,
                              _Float16* __restrict__ vT) {
  __shared__ __align__(16) _Float16 As[128 * 128];
  __shared__ __align__(16) _Float16 Bs[128 * 128];
  __shared__ float bs[128];
  int tid = threadIdx.x, wave = tid >> 5, lane = tid & 31;
  int m0 = blockIdx.x * 128;
  copy_tile_h16(As, Hh + (size_t)m0 * 128, tid);
  _Float16* outs[2] = {qh, kh};
  const float qscale = 0.08838834764831845f;  // 1/sqrt(128)
  for (int c = 0; c < 3; ++c) {
    __syncthreads();
    // wi chunk is [o][k] row-major == B^T directly
    convert_tile_f32_h16(Bs, wi + (size_t)c * 128 * 128, tid);
    if (tid < 128) bs[tid] = bi[c * 128 + tid];
    __syncthreads();
    h16 afr[4];
#pragma unroll
    for (int k = 0; k < 4; ++k)
      afr[k] = load_a(&As[(wave * 16) * 128 + k * 32], 128, lane);
    float sc = (c == 0) ? qscale : 1.f;
    for (int nt = 0; nt < 8; ++nt) {
      f8 acc = {};
#pragma unroll
      for (int k = 0; k < 4; ++k)
        acc = wmma(afr[k], load_bt(&Bs[(nt * 16) * 128 + k * 32], 128, lane), acc);
      int col = nt * 16 + (lane & 15);
#pragma unroll
      for (int v = 0; v < 8; ++v) {
        int row = m0 + wave * 16 + v + 8 * (lane >> 4);
        _Float16 hv = (_Float16)((acc[v] + bs[col]) * sc);
        if (c == 2) vT[(size_t)col * NN + row] = hv;   // V stored transposed
        else        outs[c][(size_t)row * 128 + col] = hv;
      }
    }
  }
}

// ---- Kernel 7: flash attention (Br=Bc=128, online softmax) ----------------
__global__ void flash_kernel(const _Float16* __restrict__ qh,
                             const _Float16* __restrict__ kh,
                             const _Float16* __restrict__ vT,
                             _Float16* __restrict__ oh) {
  __shared__ __align__(16) _Float16 Qs[128 * 128];
  __shared__ __align__(16) _Float16 Ks[128 * 128];
  __shared__ __align__(16) _Float16 Vs[128 * 128];  // V^T tile: [d][key]
  __shared__ __align__(16) _Float16 Ps[8][16 * 128];
  int tid = threadIdx.x, wave = tid >> 5, lane = tid & 31;
  int m0 = blockIdx.x * 128;
  copy_tile_h16(Qs, qh + (size_t)m0 * 128, tid);
  __syncthreads();
  h16 qfr[4];
#pragma unroll
  for (int k = 0; k < 4; ++k)
    qfr[k] = load_a(&Qs[(wave * 16) * 128 + k * 32], 128, lane);
  f8 o_acc[8];
  f8 zero = {};
#pragma unroll
  for (int t = 0; t < 8; ++t) o_acc[t] = zero;
  float m_run[8], l_run[8];
#pragma unroll
  for (int v = 0; v < 8; ++v) { m_run[v] = -1e30f; l_run[v] = 0.f; }

  for (int j = 0; j < NN / 128; ++j) {
    __syncthreads();
    copy_tile_h16(Ks, kh + (size_t)j * 128 * 128, tid);
    // V^T tile: row d of vT, 128 keys starting at j*128
#pragma unroll
    for (int i = tid; i < 2048; i += 256) {
      int d = i >> 4, c = i & 15;
      ((uint4*)Vs)[i] = *(const uint4*)&vT[(size_t)d * NN + j * 128 + c * 8];
    }
    __syncthreads();
    if (j + 1 < NN / 128) {
      __builtin_prefetch(&kh[(size_t)(j + 1) * 128 * 128 + lane * 64], 0, 1);
    }
    // S = Q @ K^T  (K row-major == (K^T)^T -> load_bt on K rows)
    f8 sacc[8];
    for (int nt = 0; nt < 8; ++nt) {
      f8 acc = zero;
#pragma unroll
      for (int k = 0; k < 4; ++k)
        acc = wmma(qfr[k], load_bt(&Ks[(nt * 16) * 128 + k * 32], 128, lane), acc);
      sacc[nt] = acc;
    }
    // online softmax over rows r = v + 8*(lane>>4)
    float mnew[8], scl[8], rsum[8];
#pragma unroll
    for (int v = 0; v < 8; ++v) {
      float mx = sacc[0][v];
#pragma unroll
      for (int t = 1; t < 8; ++t) mx = fmaxf(mx, sacc[t][v]);
      for (int msk = 8; msk >= 1; msk >>= 1) mx = fmaxf(mx, __shfl_xor(mx, msk, 32));
      mnew[v] = fmaxf(m_run[v], mx);
      scl[v] = __expf(m_run[v] - mnew[v]);
      rsum[v] = 0.f;
    }
#pragma unroll
    for (int t = 0; t < 8; ++t)
#pragma unroll
      for (int v = 0; v < 8; ++v) {
        float p = __expf(sacc[t][v] - mnew[v]);
        sacc[t][v] = p;
        rsum[v] += p;
      }
#pragma unroll
    for (int v = 0; v < 8; ++v) {
      for (int msk = 8; msk >= 1; msk >>= 1) rsum[v] += __shfl_xor(rsum[v], msk, 32);
      l_run[v] = l_run[v] * scl[v] + rsum[v];
      m_run[v] = mnew[v];
    }
    // P (f16) -> wave-private LDS; rescale O
    _Float16* P = &Ps[wave][0];
#pragma unroll
    for (int t = 0; t < 8; ++t) {
      int col = t * 16 + (lane & 15);
#pragma unroll
      for (int v = 0; v < 8; ++v)
        P[(v + 8 * (lane >> 4)) * 128 + col] = (_Float16)sacc[t][v];
    }
#pragma unroll
    for (int t = 0; t < 8; ++t)
#pragma unroll
      for (int v = 0; v < 8; ++v) o_acc[t][v] *= scl[v];
    // O += P @ V   (Vs holds V^T[d][key] -> load_bt rows are d)
    for (int nt = 0; nt < 8; ++nt) {
      f8 acc = o_acc[nt];
#pragma unroll
      for (int k = 0; k < 4; ++k)
        acc = wmma(load_a(&P[k * 32], 128, lane),
                   load_bt(&Vs[(nt * 16) * 128 + k * 32], 128, lane), acc);
      o_acc[nt] = acc;
    }
  }
  for (int nt = 0; nt < 8; ++nt) {
    int col = nt * 16 + (lane & 15);
#pragma unroll
    for (int v = 0; v < 8; ++v) {
      int row = m0 + wave * 16 + v + 8 * (lane >> 4);
      oh[(size_t)row * 128 + col] = (_Float16)(o_acc[nt][v] / l_run[v]);
    }
  }
}

// ---- Kernel 8: out_proj: dif = O @ wo.T + bo ------------------------------
__global__ void outproj_kernel(const _Float16* __restrict__ Ah,
                               const float* __restrict__ wo,
                               const float* __restrict__ bo,
                               float* __restrict__ out) {
  __shared__ __align__(16) _Float16 As[128 * 128];
  __shared__ __align__(16) _Float16 Bs[128 * 128];
  __shared__ float bs[128];
  int tid = threadIdx.x, wave = tid >> 5, lane = tid & 31;
  int m0 = blockIdx.x * 128;
  copy_tile_h16(As, Ah + (size_t)m0 * 128, tid);
  convert_tile_f32_h16(Bs, wo, tid);  // wo [o][k] == B^T
  if (tid < 128) bs[tid] = bo[tid];
  __syncthreads();
  h16 afr[4];
#pragma unroll
  for (int k = 0; k < 4; ++k)
    afr[k] = load_a(&As[(wave * 16) * 128 + k * 32], 128, lane);
  for (int nt = 0; nt < 8; ++nt) {
    f8 acc = {};
#pragma unroll
    for (int k = 0; k < 4; ++k)
      acc = wmma(afr[k], load_bt(&Bs[(nt * 16) * 128 + k * 32], 128, lane), acc);
    int col = nt * 16 + (lane & 15);
#pragma unroll
    for (int v = 0; v < 8; ++v) {
      int row = m0 + wave * 16 + v + 8 * (lane >> 4);
      out[(long)row * 128 + col] = acc[v] + bs[col];
    }
  }
}

// ---- Kernel 9: inter scores: mean_k tanh(sem@fcw.T+fb) @ av ---------------
__global__ void interscore_kernel(const float* __restrict__ attr,
                                  const float* __restrict__ dif,
                                  const float* __restrict__ fcw,
                                  const float* __restrict__ fcb,
                                  const float* __restrict__ av,
                                  float* __restrict__ scores) {
  __shared__ __align__(16) _Float16 A1[128 * 128];
  __shared__ __align__(16) _Float16 A2[128 * 128];
  __shared__ __align__(16) _Float16 Bs[128 * 128];
  __shared__ float fb[128], avv[128];
  int tid = threadIdx.x, wave = tid >> 5, lane = tid & 31;
  int m0 = blockIdx.x * 128;
  convert_tile_f32_h16(A1, attr + (size_t)m0 * 128, tid);
  convert_tile_f32_h16(A2, dif + (size_t)m0 * 128, tid);
  convert_tile_f32_h16(Bs, fcw, tid);  // fcw [o][k] == B^T
  if (tid < 128) { fb[tid] = fcb[tid]; avv[tid] = av[tid]; }
  __syncthreads();
  h16 a1[4], a2[4];
#pragma unroll
  for (int k = 0; k < 4; ++k) {
    a1[k] = load_a(&A1[(wave * 16) * 128 + k * 32], 128, lane);
    a2[k] = load_a(&A2[(wave * 16) * 128 + k * 32], 128, lane);
  }
  float rpart[8];
#pragma unroll
  for (int v = 0; v < 8; ++v) rpart[v] = 0.f;
  for (int nt = 0; nt < 8; ++nt) {
    f8 c1 = {}, c2 = {};
#pragma unroll
    for (int k = 0; k < 4; ++k) {
      h16 b = load_bt(&Bs[(nt * 16) * 128 + k * 32], 128, lane);
      c1 = wmma(a1[k], b, c1);
      c2 = wmma(a2[k], b, c2);
    }
    int col = nt * 16 + (lane & 15);
#pragma unroll
    for (int v = 0; v < 8; ++v) {
      float sp = 0.5f * (tanhf(c1[v] + fb[col]) + tanhf(c2[v] + fb[col]));
      rpart[v] += sp * avv[col];
    }
  }
#pragma unroll
  for (int v = 0; v < 8; ++v)
    for (int msk = 8; msk >= 1; msk >>= 1) rpart[v] += __shfl_xor(rpart[v], msk, 32);
  if ((lane & 15) == 0) {
#pragma unroll
    for (int v = 0; v < 8; ++v)
      scores[m0 + wave * 16 + v + 8 * (lane >> 4)] = rpart[v];
  }
}

// ---- Kernel 10: softmax over N + beta-weighted sum -> multi[2][T][D] ------
__global__ void interreduce_kernel(const float* __restrict__ scores,
                                   const float* __restrict__ attr,
                                   const float* __restrict__ dif,
                                   float* __restrict__ multi, int ti) {
  __shared__ float red[256];
  __shared__ float s_mx, s_sum;
  int tid = threadIdx.x;
  float mx = -1e30f;
  for (int n = tid; n < NN; n += 256) mx = fmaxf(mx, scores[n]);
  red[tid] = mx; __syncthreads();
  for (int s = 128; s >= 1; s >>= 1) {
    if (tid < s) red[tid] = fmaxf(red[tid], red[tid + s]);
    __syncthreads();
  }
  if (tid == 0) s_mx = red[0];
  __syncthreads();
  float gmx = s_mx, sm = 0.f;
  for (int n = tid; n < NN; n += 256) sm += __expf(scores[n] - gmx);
  red[tid] = sm; __syncthreads();
  for (int s = 128; s >= 1; s >>= 1) {
    if (tid < s) red[tid] += red[tid + s];
    __syncthreads();
  }
  if (tid == 0) s_sum = red[0];
  __syncthreads();
  float inv = 1.f / s_sum;
  int d = tid & 127, a = tid >> 7;  // a: 0=attr, 1=dif
  const float* src = (a == 0) ? attr : dif;
  float acc = 0.f;
  for (int n = 0; n < NN; ++n)
    acc += __expf(scores[n] - gmx) * src[(long)n * 128 + d];
  multi[(a * TT + ti) * DD + d] = acc * inv;
}

// ---- Kernel 11: final _inter over multi [2][T][D] -> out [T][D] -----------
__global__ void final_kernel(const float* __restrict__ multi,
                             const float* __restrict__ fwf,
                             const float* __restrict__ fbf,
                             const float* __restrict__ attf,
                             float* __restrict__ out) {
  __shared__ float red[128];
  __shared__ float beta_s[2];
  int d = threadIdx.x;
  float sp[2];
  for (int a = 0; a < 2; ++a) {
    float acc = 0.f;
    for (int k = 0; k < TT; ++k) {
      float dot = 0.f;
      for (int e = 0; e < 128; ++e)
        dot += multi[(a * TT + k) * DD + e] * fwf[d * 128 + e];
      acc += tanhf(dot + fbf[d]);
    }
    sp[a] = acc / (float)TT;
  }
  for (int a = 0; a < 2; ++a) {
    red[d] = sp[a] * attf[d];
    __syncthreads();
    for (int s = 64; s >= 1; s >>= 1) {
      if (d < s) red[d] += red[d + s];
      __syncthreads();
    }
    if (d == 0) beta_s[a] = red[0];
    __syncthreads();
  }
  float b0 = beta_s[0], b1 = beta_s[1];
  float m = fmaxf(b0, b1);
  float e0 = __expf(b0 - m), e1 = __expf(b1 - m);
  float w0 = e0 / (e0 + e1), w1 = e1 / (e0 + e1);
  for (int k = 0; k < TT; ++k)
    out[k * DD + d] = w0 * multi[(0 * TT + k) * DD + d] +
                      w1 * multi[(1 * TT + k) * DD + d];
}

// ---------------------------------------------------------------------------
extern "C" void kernel_launch(void* const* d_in, const int* in_sizes, int n_in,
                              void* d_out, int out_size, void* d_ws, size_t ws_size,
                              hipStream_t stream) {
  const float* x0        = (const float*)d_in[0];
  const float* xh        = (const float*)d_in[1];
  const float* att_intra = (const float*)d_in[2];
  const float* W_gcn     = (const float*)d_in[3];
  const float* b_gcn     = (const float*)d_in[4];
  const float* W_att     = (const float*)d_in[5];
  const float* b_att     = (const float*)d_in[6];
  const float* in_w      = (const float*)d_in[7];
  const float* in_b      = (const float*)d_in[8];
  const float* out_w     = (const float*)d_in[9];
  const float* out_b     = (const float*)d_in[10];
  const float* fc_w      = (const float*)d_in[11];
  const float* fc_b      = (const float*)d_in[12];
  const float* att_inter = (const float*)d_in[13];
  const float* fc_w_f    = (const float*)d_in[14];
  const float* fc_b_f    = (const float*)d_in[15];
  const float* att_f     = (const float*)d_in[16];
  const float* adj_vals  = (const float*)d_in[17];
  const float* adjt_vals = (const float*)d_in[18];
  const int*   nei       = (const int*)d_in[19];
  const int*   adj_rows  = (const int*)d_in[20];
  const int*   adj_cols  = (const int*)d_in[21];
  const int*   adjt_rows = (const int*)d_in[22];
  const int*   adjt_cols = (const int*)d_in[23];
  (void)in_sizes; (void)n_in; (void)out_size; (void)ws_size;

  char* wsp = (char*)d_ws;
  auto carve = [&](size_t bytes) {
    void* p = (void*)wsp;
    wsp += (bytes + 255) & ~(size_t)255;
    return p;
  };
  float*    one_f  = (float*)carve((size_t)NN * DD * 4);
  _Float16* one_h  = (_Float16*)carve((size_t)NN * DD * 2);
  float*    gcn_in = (float*)carve((size_t)NN * DD * 4);
  float*    att_in = (float*)carve((size_t)NN * DD * 4);
  float*    attr   = (float*)carve((size_t)NN * DD * 4);
  float*    hpre   = (float*)carve((size_t)NN * DD * 4);
  _Float16* h_h    = (_Float16*)carve((size_t)NN * DD * 2);
  _Float16* qh     = (_Float16*)carve((size_t)NN * DD * 2);
  _Float16* kh     = (_Float16*)carve((size_t)NN * DD * 2);
  _Float16* vT     = (_Float16*)carve((size_t)NN * DD * 2);  // [D][N]
  _Float16* oh     = (_Float16*)carve((size_t)NN * DD * 2);
  float*    dif    = (float*)carve((size_t)NN * DD * 4);
  float*    scores = (float*)carve((size_t)NN * 4);
  float*    multi  = (float*)carve((size_t)2 * TT * DD * 4);

  const int ROWB = NN / 128;            // 64
  const int EWB  = (EE * 32) / 256;     // 16384
  const int VECB = (NN * DD / 4) / 256; // init_bias blocks
  const int ELUB = (NN * DD / 8) / 256; // elu blocks

  for (int i = 0; i < TT; ++i) {
    intra_kernel<<<NN, 128, 0, stream>>>(x0, xh + (size_t)i * MM * DD,
                                         att_intra + i * 2 * DD,
                                         nei + (size_t)i * NN * SS, one_f, one_h);

    gemm_nn_dual<<<ROWB, 256, 0, stream>>>(one_h, W_gcn + (size_t)i * DD * DD,
                                           W_att + (size_t)i * DD * DD,
                                           gcn_in, att_in);

    init_bias<<<VECB, 256, 0, stream>>>(attr, b_gcn + i * DD);
    init_bias<<<VECB, 256, 0, stream>>>(hpre, b_att + i * DD);

    spmm_scatter<<<EWB, 256, 0, stream>>>(adjt_rows + (size_t)i * EE,
                                          adjt_cols + (size_t)i * EE,
                                          adjt_vals + (size_t)i * EE,
                                          gcn_in, attr);
    spmm_scatter<<<EWB, 256, 0, stream>>>(adj_rows + (size_t)i * EE,
                                          adj_cols + (size_t)i * EE,
                                          adj_vals + (size_t)i * EE,
                                          att_in, hpre);

    elu_to_h16<<<ELUB, 256, 0, stream>>>(hpre, h_h);

    inproj_kernel<<<ROWB, 256, 0, stream>>>(h_h, in_w + (size_t)i * 3 * DD * DD,
                                            in_b + (size_t)i * 3 * DD, qh, kh, vT);

    flash_kernel<<<ROWB, 256, 0, stream>>>(qh, kh, vT, oh);

    outproj_kernel<<<ROWB, 256, 0, stream>>>(oh, out_w + (size_t)i * DD * DD,
                                             out_b + (size_t)i * DD, dif);

    interscore_kernel<<<ROWB, 256, 0, stream>>>(attr, dif,
                                                fc_w + (size_t)i * DD * DD,
                                                fc_b + (size_t)i * DD,
                                                att_inter + (size_t)i * DD, scores);

    interreduce_kernel<<<1, 256, 0, stream>>>(scores, attr, dif, multi, i);
  }

  final_kernel<<<1, 128, 0, stream>>>(multi, fc_w_f, fc_b_f, att_f,
                                      (float*)d_out);
}
